// DynamicMemoryCell_79053168050750
// MI455X (gfx1250) — matrix operand order repaired and checked by reading, not compiled, not run on hardware
//
#include <hip/hip_runtime.h>

// Problem dims (fixed by the reference)
#define B_SZ   8192
#define IN_SZ  1024
#define MEM_SZ 1024
#define CTX_SZ 1024

typedef __bf16 bf16;
typedef bf16           v16bf  __attribute__((ext_vector_type(16)));
typedef float          v8f    __attribute__((ext_vector_type(8)));
typedef unsigned short u16v8  __attribute__((ext_vector_type(8)));
typedef unsigned short u16v16 __attribute__((ext_vector_type(16)));

__device__ __forceinline__ unsigned short f2bf(float f) {
  unsigned int u = __float_as_uint(f);
  u += 0x7FFFu + ((u >> 16) & 1u);        // round-to-nearest-even
  return (unsigned short)(u >> 16);
}

__device__ __forceinline__ float sigmoidf_(float x) { return 1.0f / (1.0f + __expf(-x)); }

__device__ __forceinline__ v8f wmma_bf16(v16bf a, v16bf b, v8f c) {
  return __builtin_amdgcn_wmma_f32_16x16x32_bf16(false, a, false, b, (short)0, c,
                                                 false, false);
}

// ---- WMMA fragment loads, direct-from-global (GRU kernel) ------------------
// A (16x32 MxK): lanes 0-15: M=lane, K=k..k+15 (contig); lanes 16-31:
// M=lane-16, K=k+16..k+31.
__device__ __forceinline__ v16bf load_a_frag(const unsigned short* A, long lda,
                                             int mbase, int kk, int lane) {
  int row = mbase + (lane & 15);
  int k0  = kk + ((lane >> 4) << 4);
  const unsigned short* p = A + (long)row * lda + k0;
  u16v8 c0 = *(const u16v8*)(p);
  u16v8 c1 = *(const u16v8*)(p + 8);
  union { u16v16 u; v16bf v; } r;
#pragma unroll
  for (int i = 0; i < 8; ++i) { r.u[i] = c0[i]; r.u[8 + i] = c1[i]; }
  return r.v;
}

// B (32x16 KxN): lanes 0-15: col n=lane, K=kk..kk+15 (contig); lanes 16-31:
// col n=lane-16, K=kk+16..kk+31. W is [N,K] row-major (computes x @ W^T).
__device__ __forceinline__ v16bf load_b_frag(const unsigned short* W, long ldw,
                                             int nbase, int kk, int lane) {
  int col = nbase + (lane & 15);
  int kb  = kk + ((lane >> 4) << 4);
  union { u16v16 u; v16bf v; } r;
  r.u = *(const u16v16*)(W + (long)col * ldw + kb);
  return r.v;
}

// ---- WMMA fragment loads from padded LDS tiles -----------------------------
#define LDP 40  // padded row stride (shorts): 80B = 20 banks; 16B-aligned rows
__device__ __forceinline__ v16bf lds_a_frag(const unsigned short* sA, int mrow,
                                            int lane) {
  int row = mrow * 16 + (lane & 15);
  int k0  = (lane >> 4) << 4;
  const unsigned short* p = sA + row * LDP + k0;
  u16v8 c0 = *(const u16v8*)(p);
  u16v8 c1 = *(const u16v8*)(p + 8);
  union { u16v16 u; v16bf v; } r;
#pragma unroll
  for (int i = 0; i < 8; ++i) { r.u[i] = c0[i]; r.u[8 + i] = c1[i]; }
  return r.v;
}

__device__ __forceinline__ v16bf lds_b_frag(const unsigned short* sB, int ncol,
                                            int lane) {
  int col = ncol + (lane & 15);
  int kb  = (lane >> 4) << 4;
  union { u16v16 u; v16bf v; } r;
  r.u = *(const u16v16*)(sB + col * LDP + kb);
  return r.v;
}

// ---- LDS-staged GEMM: D[M,N] = A[M,K](bf16,lda) x W[N,K]^T(bf16) + bias ----
// Block tile 64x128, K-step 32, double-buffered LDS (one barrier per step);
// 8 waves = 4 m-tiles x 2 n-groups; each wave computes a 16x64 strip
// (4 accumulators, A fragment reused 4x from registers).
// EPI 0: store bf16(D)
// EPI 1: g = sigmoid(D); gm = aux*g; store f32 gm and bf16 gm   (forget gate)
// EPI 2: s = sigmoid(D); store f32 s*aux                         (output gate)
template <int EPI>
__global__ void __launch_bounds__(256)
gemm_lds_k(const unsigned short* __restrict__ A, long lda,
           const unsigned short* __restrict__ W, int K,
           const float* __restrict__ bias, int M, int N,
           unsigned short* __restrict__ obf, float* __restrict__ of32,
           const float* __restrict__ aux) {
  __shared__ unsigned short sA[2][64 * LDP];    // 2 x 5120 B
  __shared__ unsigned short sB[2][128 * LDP];   // 2 x 10240 B

  int nb   = N >> 7;                          // N / 128
  int mblk = ((int)blockIdx.x / nb) << 6;     // * 64
  int nblk = ((int)blockIdx.x % nb) << 7;     // * 128
  int t    = threadIdx.x;
  int lane = t & 31;
  int wv   = t >> 5;
  int mrow = wv & 3;                          // m-tile within block
  int ngrp = wv >> 2;                         // 0/1 -> 64-wide n group

  int arow = t >> 2, aseg = t & 3;            // A: 64 rows x 4 segs of 8
  int brow = t >> 1, bseg = t & 1;            // B: 128 rows x 2 segs of 16
  const unsigned short* gA = A + (long)(mblk + arow) * lda + aseg * 8;
  const unsigned short* gB = W + (long)(nblk + brow) * (long)K + bseg * 16;
  int aoff = arow * LDP + aseg * 8;
  int boff = brow * LDP + bseg * 16;

  // prologue: stage K-slice 0 into buffer 0
  *(u16v8*)(&sA[0][aoff])  = *(const u16v8*)(gA);
  *(u16v16*)(&sB[0][boff]) = *(const u16v16*)(gB);
  __syncthreads();

  v8f acc[4] = {};
  int p = 0;
  for (int kk = 0; kk < K; kk += 32) {
    int kn = kk + 32;
    u16v8  na = {};
    u16v16 nbv = {};
    if (kn < K) {                             // uniform (scalar) branch
      na  = *(const u16v8*)(gA + kn);
      nbv = *(const u16v16*)(gB + kn);
    }
    v16bf a = lds_a_frag(sA[p], mrow, lane);
#pragma unroll
    for (int tt = 0; tt < 4; ++tt) {
      v16bf b = lds_b_frag(sB[p], ngrp * 64 + tt * 16, lane);
      acc[tt] = wmma_bf16(a, b, acc[tt]);
    }
    if (kn < K) {                             // write the other buffer
      *(u16v8*)(&sA[p ^ 1][aoff])  = na;
      *(u16v16*)(&sB[p ^ 1][boff]) = nbv;
    }
    __syncthreads();                          // one barrier per K-step
    p ^= 1;
  }

  int lr = lane & 15, lh = lane >> 4;
#pragma unroll
  for (int tt = 0; tt < 4; ++tt) {
    int  nn = nblk + ngrp * 64 + tt * 16 + lr;
    float bvv = bias[nn];
#pragma unroll
    for (int v = 0; v < 8; ++v) {
      int  m   = mblk + mrow * 16 + v + 8 * lh;
      long idx = (long)m * N + nn;
      float c  = acc[tt][v] + bvv;
      if (EPI == 0) {
        obf[idx] = f2bf(c);
      } else if (EPI == 1) {
        float gm = aux[idx] * sigmoidf_(c);
        of32[idx] = gm;
        obf[idx]  = f2bf(gm);
      } else {
        of32[idx] = sigmoidf_(c) * aux[idx];
      }
    }
  }
}

// ---- Fused GRU cell: 6 WMMA accumulators, software-pipelined fragments -----
__global__ void __launch_bounds__(256)
gru_fused_k(const unsigned short* __restrict__ Actx,   // [B,CTX] bf16
            const unsigned short* __restrict__ Agat,   // [B,MEM] bf16
            const float* __restrict__ gated_f32,       // [B,MEM]
            const unsigned short* __restrict__ Wih,    // [3*MEM,CTX] bf16
            const float* __restrict__ bih,
            const unsigned short* __restrict__ Whh,    // [3*MEM,MEM] bf16
            const float* __restrict__ bhh,
            unsigned short* __restrict__ upd_bf, float* __restrict__ upd_f32,
            int M, int N, int K) {
  int wave = (int)((blockIdx.x * blockDim.x + threadIdx.x) >> 5);
  int lane = threadIdx.x & 31;
  int ntiles = N >> 4;
  int mt = (wave / ntiles) << 4;
  int nt = (wave % ntiles) << 4;
  if (mt >= M) return;                        // wave-uniform; EXEC all-ones

  long gs   = (long)N * K;                    // per-gate row block in Wih/Whh
  int  wcol = nt + (lane & 15);               // weight row touched by this lane

  v8f air = {}, aiz = {}, ain = {}, ahr = {}, ahz = {}, ahn = {};

  // prologue: fragment generation 0
  v16bf ai = load_a_frag(Actx, K, mt, 0, lane);
  v16bf ah = load_a_frag(Agat, K, mt, 0, lane);
  v16bf b0 = load_b_frag(Wih,          K, nt, 0, lane);
  v16bf b1 = load_b_frag(Wih + gs,     K, nt, 0, lane);
  v16bf b2 = load_b_frag(Wih + 2 * gs, K, nt, 0, lane);
  v16bf b3 = load_b_frag(Whh,          K, nt, 0, lane);
  v16bf b4 = load_b_frag(Whh + gs,     K, nt, 0, lane);
  v16bf b5 = load_b_frag(Whh + 2 * gs, K, nt, 0, lane);

  // steady state: load generation kk while 6 WMMAs consume generation kk-32
  for (int kk = 32; kk < K; kk += 32) {
    v16bf nai = load_a_frag(Actx, K, mt, kk, lane);
    v16bf nah = load_a_frag(Agat, K, mt, kk, lane);
    v16bf nb0 = load_b_frag(Wih,          K, nt, kk, lane);
    v16bf nb1 = load_b_frag(Wih + gs,     K, nt, kk, lane);
    v16bf nb2 = load_b_frag(Wih + 2 * gs, K, nt, kk, lane);
    v16bf nb3 = load_b_frag(Whh,          K, nt, kk, lane);
    v16bf nb4 = load_b_frag(Whh + gs,     K, nt, kk, lane);
    v16bf nb5 = load_b_frag(Whh + 2 * gs, K, nt, kk, lane);
    // keep next K-slices of the weight streams in near caches
    __builtin_prefetch(Wih + (long)wcol * K + kk + 96, 0, 3);
    __builtin_prefetch(Whh + (long)wcol * K + kk + 96, 0, 3);
    air = wmma_bf16(ai, b0, air);
    aiz = wmma_bf16(ai, b1, aiz);
    ain = wmma_bf16(ai, b2, ain);
    ahr = wmma_bf16(ah, b3, ahr);
    ahz = wmma_bf16(ah, b4, ahz);
    ahn = wmma_bf16(ah, b5, ahn);
    ai = nai; ah = nah;
    b0 = nb0; b1 = nb1; b2 = nb2; b3 = nb3; b4 = nb4; b5 = nb5;
  }
  // tail
  air = wmma_bf16(ai, b0, air);
  aiz = wmma_bf16(ai, b1, aiz);
  ain = wmma_bf16(ai, b2, ain);
  ahr = wmma_bf16(ah, b3, ahr);
  ahz = wmma_bf16(ah, b4, ahz);
  ahn = wmma_bf16(ah, b5, ahn);

  int lr = lane & 15, lh = lane >> 4;
  int nn = nt + lr;
  float br = bih[nn], bz = bih[N + nn], bn = bih[2 * N + nn];
  float hr = bhh[nn], hz = bhh[N + nn], hn = bhh[2 * N + nn];
#pragma unroll
  for (int v = 0; v < 8; ++v) {
    int  m   = mt + v + 8 * lh;
    long idx = (long)m * N + nn;
    float r  = sigmoidf_(air[v] + br + ahr[v] + hr);
    float z  = sigmoidf_(aiz[v] + bz + ahz[v] + hz);
    float nf = tanhf(ain[v] + bn + r * (ahn[v] + hn));
    float gm = gated_f32[idx];
    float u  = (1.0f - z) * nf + z * gm;
    upd_f32[idx] = u;
    upd_bf[idx]  = f2bf(u);
  }
}

// ---- Vectorized fp32 -> bf16 converts (b128 in, b128 out) ------------------
__global__ void cvt8_k(const float* __restrict__ src,
                       unsigned short* __restrict__ dst, long n8) {
  long i = (long)blockIdx.x * blockDim.x + threadIdx.x;
  if (i < n8) {
    const float4* s = (const float4*)src + 2 * i;
    float4 a = s[0], b = s[1];
    u16v8 o;
    o[0] = f2bf(a.x); o[1] = f2bf(a.y); o[2] = f2bf(a.z); o[3] = f2bf(a.w);
    o[4] = f2bf(b.x); o[5] = f2bf(b.y); o[6] = f2bf(b.z); o[7] = f2bf(b.w);
    *(u16v8*)(dst + 8 * i) = o;
  }
}

// src [rows,cols] contiguous -> dst rows with leading dim dld at column coff
__global__ void cvt8_strided_k(const float* __restrict__ src,
                               unsigned short* __restrict__ dst, int rows,
                               int cols, int dld, int coff) {
  long i = (long)blockIdx.x * blockDim.x + threadIdx.x;
  int  c8 = cols >> 3;
  if (i < (long)rows * c8) {
    int r = (int)(i / c8), c = (int)(i % c8);
    const float4* s = (const float4*)(src + (long)r * cols + c * 8);
    float4 a = s[0], b = s[1];
    u16v8 o;
    o[0] = f2bf(a.x); o[1] = f2bf(a.y); o[2] = f2bf(a.z); o[3] = f2bf(a.w);
    o[4] = f2bf(b.x); o[5] = f2bf(b.y); o[6] = f2bf(b.z); o[7] = f2bf(b.w);
    *(u16v8*)(dst + (long)r * dld + coff + c * 8) = o;
  }
}

__global__ void ones_k(float* __restrict__ p, int n) {
  int i = blockIdx.x * blockDim.x + threadIdx.x;
  if (i < n) p[i] = 1.0f;
}

// ---- Host-side orchestration ----------------------------------------------
extern "C" void kernel_launch(void* const* d_in, const int* in_sizes, int n_in,
                              void* d_out, int out_size, void* d_ws, size_t ws_size,
                              hipStream_t stream) {
  const float* input   = (const float*)d_in[0];   // [B,IN]
  const float* prevmem = (const float*)d_in[1];   // [B,MEM]
  const float* inprj_w = (const float*)d_in[2];   // [3*CTX,CTX]
  const float* inprj_b = (const float*)d_in[3];
  const float* outp_w  = (const float*)d_in[4];   // [CTX,CTX]
  const float* outp_b  = (const float*)d_in[5];
  const float* ip_w    = (const float*)d_in[6];   // [CTX,IN]
  const float* ip_b    = (const float*)d_in[7];
  // d_in[8]=mp_w, d_in[9]=mp_b: dead (only feed k; softmax over 1 key == 1)
  const float* fg_w    = (const float*)d_in[10];  // [MEM,MEM+IN]
  const float* fg_b    = (const float*)d_in[11];
  const float* og_w    = (const float*)d_in[12];  // [MEM,MEM]
  const float* og_b    = (const float*)d_in[13];
  const float* gw_ih   = (const float*)d_in[14];  // [3*MEM,CTX]
  const float* gb_ih   = (const float*)d_in[15];
  const float* gw_hh   = (const float*)d_in[16];  // [3*MEM,MEM]
  const float* gb_hh   = (const float*)d_in[17];

  const float* wv = inprj_w + (long)2 * CTX_SZ * CTX_SZ;  // v slice of in_proj
  const float* bv = inprj_b + 2 * CTX_SZ;

  // workspace carve-out (bf16 stored as u16)
  unsigned char* wp = (unsigned char*)d_ws;
  auto take = [&](size_t bytes) -> void* {
    void* q = wp;
    wp += (bytes + 255) & ~(size_t)255;
    return q;
  };
  unsigned short* cat    = (unsigned short*)take((size_t)B_SZ * 2048 * 2); // [prev|input]
  unsigned short* pi     = (unsigned short*)take((size_t)B_SZ * CTX_SZ * 2);
  unsigned short* vbuf   = (unsigned short*)take((size_t)B_SZ * CTX_SZ * 2);
  unsigned short* ctx    = (unsigned short*)take((size_t)B_SZ * CTX_SZ * 2);
  unsigned short* gat_bf = (unsigned short*)take((size_t)B_SZ * MEM_SZ * 2);
  float*          gat_f  = (float*)take((size_t)B_SZ * MEM_SZ * 4);
  unsigned short* upd_bf = (unsigned short*)take((size_t)B_SZ * MEM_SZ * 2);
  float*          upd_f  = (float*)take((size_t)B_SZ * MEM_SZ * 4);
  unsigned short* w_ip = (unsigned short*)take((size_t)CTX_SZ * IN_SZ * 2);
  unsigned short* w_v  = (unsigned short*)take((size_t)CTX_SZ * CTX_SZ * 2);
  unsigned short* w_op = (unsigned short*)take((size_t)CTX_SZ * CTX_SZ * 2);
  unsigned short* w_fg = (unsigned short*)take((size_t)MEM_SZ * 2048 * 2);
  unsigned short* w_ih = (unsigned short*)take((size_t)3 * MEM_SZ * CTX_SZ * 2);
  unsigned short* w_hh = (unsigned short*)take((size_t)3 * MEM_SZ * MEM_SZ * 2);
  unsigned short* w_og = (unsigned short*)take((size_t)MEM_SZ * MEM_SZ * 2);

  float* out      = (float*)d_out;                    // [B,MEM]
  float* out_attn = out + (long)B_SZ * MEM_SZ;        // [B,1,1]

  const int T = 256;
  auto blk = [](long n, int t) { return (unsigned)((n + t - 1) / t); };

  // 1) bf16 conversions (vectorized, 8 elems/thread)
  cvt8_strided_k<<<blk((long)B_SZ * MEM_SZ / 8, T), T, 0, stream>>>(
      prevmem, cat, B_SZ, MEM_SZ, 2048, 0);
  cvt8_strided_k<<<blk((long)B_SZ * IN_SZ / 8, T), T, 0, stream>>>(
      input, cat, B_SZ, IN_SZ, 2048, MEM_SZ);
  cvt8_k<<<blk((long)CTX_SZ * IN_SZ / 8, T), T, 0, stream>>>(
      ip_w, w_ip, (long)CTX_SZ * IN_SZ / 8);
  cvt8_k<<<blk((long)CTX_SZ * CTX_SZ / 8, T), T, 0, stream>>>(
      wv, w_v, (long)CTX_SZ * CTX_SZ / 8);
  cvt8_k<<<blk((long)CTX_SZ * CTX_SZ / 8, T), T, 0, stream>>>(
      outp_w, w_op, (long)CTX_SZ * CTX_SZ / 8);
  cvt8_k<<<blk((long)MEM_SZ * 2048 / 8, T), T, 0, stream>>>(
      fg_w, w_fg, (long)MEM_SZ * 2048 / 8);
  cvt8_k<<<blk((long)3 * MEM_SZ * CTX_SZ / 8, T), T, 0, stream>>>(
      gw_ih, w_ih, (long)3 * MEM_SZ * CTX_SZ / 8);
  cvt8_k<<<blk((long)3 * MEM_SZ * MEM_SZ / 8, T), T, 0, stream>>>(
      gw_hh, w_hh, (long)3 * MEM_SZ * MEM_SZ / 8);
  cvt8_k<<<blk((long)MEM_SZ * MEM_SZ / 8, T), T, 0, stream>>>(
      og_w, w_og, (long)MEM_SZ * MEM_SZ / 8);

  // LDS-staged GEMM grid: 64x128 block tiles
  unsigned gblocks = (unsigned)((B_SZ / 64) * (1024 / 128));   // 1024

  // 2) pi = x @ ip_w^T + b        (A = input half of cat, lda=2048)
  gemm_lds_k<0><<<gblocks, T, 0, stream>>>(cat + MEM_SZ, 2048, w_ip, IN_SZ, ip_b,
                                           B_SZ, CTX_SZ, pi, nullptr, nullptr);
  // 3) v = pi @ wv^T + bv
  gemm_lds_k<0><<<gblocks, T, 0, stream>>>(pi, CTX_SZ, w_v, CTX_SZ, bv, B_SZ,
                                           CTX_SZ, vbuf, nullptr, nullptr);
  // 4) ctx = v @ out_proj^T + b   (attn == v since softmax over 1 key)
  gemm_lds_k<0><<<gblocks, T, 0, stream>>>(vbuf, CTX_SZ, w_op, CTX_SZ, outp_b,
                                           B_SZ, CTX_SZ, ctx, nullptr, nullptr);
  // 5) gated = prev_mem * sigmoid([prev|x] @ fg_w^T + b)   (K = 2048)
  gemm_lds_k<1><<<gblocks, T, 0, stream>>>(cat, 2048, w_fg, 2048, fg_b, B_SZ,
                                           MEM_SZ, gat_bf, gat_f, prevmem);
  // 6) fused GRU cell -> updated (6 accumulators; gi/gh never materialized)
  unsigned gru_blocks = (unsigned)(((long)(B_SZ / 16) * (1024 / 16)) / 8); // 4096
  gru_fused_k<<<gru_blocks, T, 0, stream>>>(ctx, gat_bf, gat_f, w_ih, gb_ih, w_hh,
                                            gb_hh, upd_bf, upd_f, B_SZ, MEM_SZ,
                                            CTX_SZ);
  // 7) out = updated * sigmoid(updated @ og_w^T + b)
  gemm_lds_k<2><<<gblocks, T, 0, stream>>>(upd_bf, MEM_SZ, w_og, MEM_SZ, og_b,
                                           B_SZ, MEM_SZ, nullptr, out, upd_f);
  // 8) attention weights are exactly 1.0 (softmax over a single key)
  ones_k<<<blk(B_SZ, T), T, 0, stream>>>(out_attn, B_SZ);

  (void)in_sizes; (void)n_in; (void)out_size; (void)ws_size;
}
